// GLeMaNet_55851754717509
// MI455X (gfx1250) — matrix operand order repaired and checked by reading, not compiled, not run on hardware
//
#include <hip/hip_runtime.h>
#include <math.h>

typedef __attribute__((ext_vector_type(16))) _Float16 v16h;
typedef __attribute__((ext_vector_type(8)))  float    v8f;
typedef __attribute__((ext_vector_type(8)))  unsigned int v8u;

#define BB  8
#define NB  1024
#define DD  128
#define HH  2
#define LL  4
#define DFCN 512

#define PADK 36   // padded K-row length in halves (72 B: dword & 8B aligned rows)

static __device__ __forceinline__ unsigned pack2(float x, float y)
{
    union { _Float16 h[2]; unsigned u; } p;
    p.h[0] = (_Float16)x; p.h[1] = (_Float16)y;
    return p.u;
}

// ---------------------------------------------------------------------------
// WMMA GEMM: C[M,N] = act(A[M,K] @ opB(B)), row-major fp32 in/out, f16 WMMA
// with fp32 accumulation.
// Block = 256 threads (8 waves, 2x4 wave grid). Block tile 64(M) x 128(N).
// Each wave owns a 32x32 output tile: 2 A-frags x 2 B-frags -> 4 independent
// v_wmma_f32_16x16x32_f16 per K-step. LDS tiles stored so every WMMA operand
// half-pair is one aligned dword -> fragment gather = 8x ds_load_b32.
// Requires M%64==0, N%128==0, K%32==0 (true for all call sites).
// transB: B is [N,K] and we multiply by B^T. Batched via blockIdx.z strides.
// ---------------------------------------------------------------------------
__global__ __launch_bounds__(256)
void k_gemm_wmma(const float* __restrict__ A, const float* __restrict__ Bm,
                 float* __restrict__ C,
                 int K, int lda, int ldb, int ldc,
                 long long sA, long long sB, long long sC,
                 int transB, int doRelu)
{
    __shared__ _Float16 As[64][PADK];    // A tile:  M64 x K32 (row-major)
    __shared__ _Float16 Bt[128][PADK];   // B tile transposed: N128 x K32

    const int tid  = threadIdx.x;
    const int lane = tid & 31;
    const int wave = tid >> 5;
    const int wm   = wave & 1;           // M subtile (2 x 32)
    const int wn   = wave >> 1;          // N subtile (4 x 32)
    const int rowBase = blockIdx.y * 64;
    const int colBase = blockIdx.x * 128;
    const int bat = blockIdx.z;

    const float* Ab = A  + (long long)bat * sA;
    const float* Bb = Bm + (long long)bat * sB;
    float*       Cb = C  + (long long)bat * sC;

    const int mr   = lane & 15;
    const int half = lane >> 4;

    v8f acc[2][2] = {};

    for (int kBase = 0; kBase < K; kBase += 32) {
        // ---- A tile: 64x32 fp32 = 512 float4; 2 per thread ----
        #pragma unroll
        for (int i = 0; i < 2; ++i) {
            int id = tid + 256 * i;
            int m  = id >> 3;
            int k4 = id & 7;
            const float* src = Ab + (long long)(rowBase + m) * lda + kBase + k4 * 4;
            if (kBase + 32 < K) __builtin_prefetch(src + 32, 0, 3);
            const float4 f = *(const float4*)src;
            unsigned* dst = (unsigned*)&As[m][0];
            dst[k4 * 2]     = pack2(f.x, f.y);
            dst[k4 * 2 + 1] = pack2(f.z, f.w);
        }
        // ---- B tile -> transposed LDS layout Bt[n][k] ----
        if (transB) {
            // B[N,K]: 128 rows x 8 float4; 4 per thread; contiguous stores
            #pragma unroll
            for (int i = 0; i < 4; ++i) {
                int id = tid + 256 * i;
                int n  = id >> 3;
                int k4 = id & 7;
                const float* src = Bb + (long long)(colBase + n) * ldb + kBase + k4 * 4;
                if (kBase + 32 < K) __builtin_prefetch(src + 32, 0, 3);
                const float4 f = *(const float4*)src;
                unsigned* dst = (unsigned*)&Bt[n][0];
                dst[k4 * 2]     = pack2(f.x, f.y);
                dst[k4 * 2 + 1] = pack2(f.z, f.w);
            }
        } else {
            // B[K,N]: 32 rows x 32 float4; 4 per thread; scatter 4 cols
            #pragma unroll
            for (int i = 0; i < 4; ++i) {
                int id = tid + 256 * i;
                int kk = id >> 5;
                int n4 = (id & 31) * 4;
                const float* src = Bb + (long long)(kBase + kk) * ldb + colBase + n4;
                if (kBase + 32 < K) __builtin_prefetch(src + (long long)32 * ldb, 0, 3);
                const float4 f = *(const float4*)src;
                Bt[n4 + 0][kk] = (_Float16)f.x;
                Bt[n4 + 1][kk] = (_Float16)f.y;
                Bt[n4 + 2][kk] = (_Float16)f.z;
                Bt[n4 + 3][kk] = (_Float16)f.w;
            }
        }
        __syncthreads();

        // ---- gather fragments: 8x ds_load_b32 each (ISA 7.12.2 layouts) ----
        // A 16x32 f16: K pair index = (v/4)*8 + half*4 + (v%4)
        // B 32x16 f16: K pair index = (v/4)*8 + half*4 + (v%4)
        v16h afrag[2], bfrag[2];
        #pragma unroll
        for (int i = 0; i < 2; ++i) {
            const unsigned* arow = (const unsigned*)&As[wm * 32 + i * 16 + mr][0];
            const unsigned* brow = (const unsigned*)&Bt[wn * 32 + i * 16 + mr][0];
            v8u au, bu;
            #pragma unroll
            for (int v = 0; v < 8; ++v) {
                au[v] = arow[(v >> 2) * 8 + half * 4 + (v & 3)];
                bu[v] = brow[(v >> 2) * 8 + half * 4 + (v & 3)];
            }
            afrag[i] = __builtin_bit_cast(v16h, au);
            bfrag[i] = __builtin_bit_cast(v16h, bu);
        }

        #pragma unroll
        for (int i = 0; i < 2; ++i)
            #pragma unroll
            for (int j = 0; j < 2; ++j)
                acc[i][j] = __builtin_amdgcn_wmma_f32_16x16x32_f16(
                    false, afrag[i], false, bfrag[j], (short)0, acc[i][j],
                    false, false);
        __syncthreads();
    }

    // ---- store C per 16x16 f32 C/D layout ----
    const int roff = half * 8;
    #pragma unroll
    for (int i = 0; i < 2; ++i)
        #pragma unroll
        for (int j = 0; j < 2; ++j)
            #pragma unroll
            for (int r = 0; r < 8; ++r) {
                float v = acc[i][j][r];
                if (doRelu) v = fmaxf(v, 0.f);
                Cb[(long long)(rowBase + wm * 32 + i * 16 + roff + r) * ldc +
                   colBase + wn * 32 + j * 16 + mr] = v;
            }
}

// ---------------------------------------------------------------------------
// [B,N,H*D] -> [B,H,N,D]
// ---------------------------------------------------------------------------
__global__ void k_permute_h(const float* __restrict__ hlin, float* __restrict__ h)
{
    int idx = blockIdx.x * blockDim.x + threadIdx.x;   // over B*H*N*D = 2M
    if (idx >= BB * HH * NB * DD) return;
    int d  = idx & (DD - 1);
    int n  = (idx >> 7) & (NB - 1);
    int hd = (idx >> 17) & (HH - 1);
    int b  = idx >> 18;
    h[idx] = hlin[((long long)(b * NB + n)) * (HH * DD) + hd * DD + d];
}

// ---------------------------------------------------------------------------
// In-place symmetrize: e[j,k] = e[k,j] = e[j,k] + e[k,j]  (per bh batch)
// ---------------------------------------------------------------------------
__global__ void k_esym(float* __restrict__ e)
{
    int k = blockIdx.x * 16 + threadIdx.x;
    int j = blockIdx.y * 16 + threadIdx.y;
    if (j > k) return;
    float* eb = e + (long long)blockIdx.z * NB * NB;
    float s = eb[(long long)j * NB + k] + eb[(long long)k * NB + j];
    eb[(long long)j * NB + k] = s;
    eb[(long long)k * NB + j] = s;
}

// ---------------------------------------------------------------------------
// Column softmax (axis=-2) with multiplicative pre-mask and post-scale by adj.
// One thread per column k; coalesced across threads.
// ---------------------------------------------------------------------------
__global__ void k_masked_softmax(const float* __restrict__ esym,
                                 const float* __restrict__ adj,
                                 float* __restrict__ att)
{
    const int k  = blockIdx.x * blockDim.x + threadIdx.x;
    const int bh = blockIdx.y;
    const int b  = bh >> 1;
    const float* ecol = esym + (long long)bh * NB * NB + k;
    const float* acol = adj  + (long long)b  * NB * NB + k;
    float*       ocol = att  + (long long)bh * NB * NB + k;

    float m = -1e30f;
    for (int j = 0; j < NB; ++j) {
        float aj = acol[(long long)j * NB];
        float p  = (aj > 0.f) ? ecol[(long long)j * NB] : 0.f;
        m = fmaxf(m, p);
    }
    float s = 0.f;
    for (int j = 0; j < NB; ++j) {
        float aj = acol[(long long)j * NB];
        float p  = (aj > 0.f) ? ecol[(long long)j * NB] : 0.f;
        s += __expf(p - m);
    }
    float inv = 1.f / s;
    for (int j = 0; j < NB; ++j) {
        float aj = acol[(long long)j * NB];
        float p  = (aj > 0.f) ? ecol[(long long)j * NB] : 0.f;
        ocol[(long long)j * NB] = __expf(p - m) * inv * aj;
    }
}

// ---------------------------------------------------------------------------
// beta = sigmoid(concat(h, az) @ Wbw + Wbb), one thread per (b,h,n) row
// ---------------------------------------------------------------------------
__global__ void k_beta(const float* __restrict__ h, const float* __restrict__ az,
                       const float* __restrict__ wbw, const float* __restrict__ wbb,
                       float* __restrict__ beta)
{
    int i = blockIdx.x * blockDim.x + threadIdx.x;
    if (i >= BB * HH * NB) return;
    const float* hr = h  + (long long)i * DD;
    const float* ar = az + (long long)i * DD;
    float s = wbb[0];
    for (int d = 0; d < DD; ++d) s += hr[d] * wbw[d] + ar[d] * wbw[DD + d];
    beta[i] = 1.f / (1.f + __expf(-s));
}

__global__ void k_zupdate(const float* __restrict__ h, const float* __restrict__ az,
                          const float* __restrict__ beta, float* __restrict__ z)
{
    int idx = blockIdx.x * blockDim.x + threadIdx.x;
    if (idx >= BB * HH * NB * DD) return;
    float bt = beta[idx >> 7];
    z[idx] = bt * h[idx] + (1.f - bt) * az[idx];
}

// ---------------------------------------------------------------------------
// [B,H,N,D] -> relu -> [B,N,H*D]
// ---------------------------------------------------------------------------
__global__ void k_zcat_relu(const float* __restrict__ z, float* __restrict__ zcat)
{
    int idx = blockIdx.x * blockDim.x + threadIdx.x;
    if (idx >= BB * HH * NB * DD) return;
    int d  = idx & (DD - 1);
    int n  = (idx >> 7) & (NB - 1);
    int hd = (idx >> 17) & (HH - 1);
    int b  = idx >> 18;
    zcat[((long long)(b * NB + n)) * (HH * DD) + hd * DD + d] = fmaxf(z[idx], 0.f);
}

__global__ void k_combine(const float* __restrict__ c2, const float* __restrict__ c1,
                          float* __restrict__ c)
{
    int idx = blockIdx.x * blockDim.x + threadIdx.x;
    if (idx >= BB * NB * DD) return;
    c[idx] = c2[idx] - c1[idx];
}

// ---------------------------------------------------------------------------
// Masked mean over nodes: pooled[b,d] = sum_n c*valid / sum_n valid
// ---------------------------------------------------------------------------
__global__ void k_pool(const float* __restrict__ c, const float* __restrict__ valid,
                       float* __restrict__ pooled)
{
    int b = blockIdx.x, d = threadIdx.x;
    float s = 0.f, vs = 0.f;
    for (int n = 0; n < NB; ++n) {
        float v = valid[b * NB + n];
        s  += c[((long long)b * NB + n) * DD + d] * v;
        vs += v;
    }
    pooled[b * DD + d] = s / vs;
}

// ---------------------------------------------------------------------------
// Tiny FC head (M=8): act 0=none 1=relu 2=sigmoid
// ---------------------------------------------------------------------------
__global__ void k_fc(const float* __restrict__ in, const float* __restrict__ w,
                     const float* __restrict__ bias, float* __restrict__ out,
                     int M, int K, int N, int act)
{
    int idx = blockIdx.x * blockDim.x + threadIdx.x;
    if (idx >= M * N) return;
    int m = idx / N, n = idx % N;
    float s = bias[n];
    for (int kk = 0; kk < K; ++kk) s += in[m * K + kk] * w[kk * N + n];
    if (act == 1) s = fmaxf(s, 0.f);
    else if (act == 2) s = 1.f / (1.f + __expf(-s));
    out[idx] = s;
}

// ---------------------------------------------------------------------------
extern "C" void kernel_launch(void* const* d_in, const int* in_sizes, int n_in,
                              void* d_out, int out_size, void* d_ws, size_t ws_size,
                              hipStream_t stream)
{
    (void)in_sizes; (void)n_in; (void)out_size; (void)ws_size;

    const float* x       = (const float*)d_in[0];
    const float* adj1    = (const float*)d_in[1];
    const float* adj2    = (const float*)d_in[2];
    const float* valid   = (const float*)d_in[3];
    const float* embede  = (const float*)d_in[4];
    const float* Wh      = (const float*)d_in[5];   // [L,128,256]
    const float* We      = (const float*)d_in[6];   // [L,128,128]
    const float* Wbw     = (const float*)d_in[7];   // [L,256,1]
    const float* Wbb     = (const float*)d_in[8];   // [L,1]
    const float* Wo      = (const float*)d_in[9];   // [L,256,128]
    const float* fc0_w   = (const float*)d_in[10];
    const float* fc0_b   = (const float*)d_in[11];
    const float* fc1_w   = (const float*)d_in[12];
    const float* fc1_b   = (const float*)d_in[13];
    const float* fc2_w   = (const float*)d_in[14];
    const float* fc2_b   = (const float*)d_in[15];

    // workspace carve-up (float units)
    float* ws = (float*)d_ws;
    size_t off = 0;
    auto alloc = [&](size_t nf) { float* p = ws + off; off += nf; return p; };
    const size_t SZ_C  = (size_t)BB * NB * DD;          // 1M
    const size_t SZ_H  = (size_t)BB * HH * NB * DD;     // 2M
    const size_t SZ_E  = (size_t)BB * HH * NB * NB;     // 16M

    float* c      = alloc(SZ_C);
    float* c1buf  = alloc(SZ_C);
    float* c2buf  = alloc(SZ_C);
    float* hlin   = alloc(SZ_H);
    float* h      = alloc(SZ_H);
    float* hW     = alloc(SZ_H);
    float* e      = alloc(SZ_E);
    float* att    = alloc(SZ_E);
    float* z      = alloc(SZ_H);
    float* az     = alloc(SZ_H);
    float* beta   = alloc((size_t)BB * HH * NB);
    float* zcat   = alloc(SZ_H);
    float* pooled = alloc((size_t)BB * DD);
    float* f0     = alloc((size_t)BB * DFCN);
    float* f1     = alloc((size_t)BB * DFCN);

    auto gemm = [&](const float* A, const float* B, float* C,
                    int M, int N, int K, int lda, int ldb, int ldc,
                    long long sA, long long sB, long long sC,
                    int batch, int transB, int relu) {
        dim3 g(N / 128, M / 64, batch), blk(256);
        k_gemm_wmma<<<g, blk, 0, stream>>>(A, B, C, K, lda, ldb, ldc,
                                           sA, sB, sC, transB, relu);
    };

    const int ELT = 256;
    const int nH = (int)SZ_H, nC = (int)SZ_C;

    // c = x @ embede_w
    gemm(x, embede, c, BB * NB, DD, DD, DD, DD, DD, 0, 0, 0, 1, 0, 0);

    const float* adjs[2] = { adj1, adj2 };
    float* couts[2] = { c1buf, c2buf };

    for (int k = 0; k < LL; ++k) {
        // h = relu(c @ Wh[k]) : [8192,128]@[128,256]
        gemm(c, Wh + (size_t)k * DD * DD * HH, hlin,
             BB * NB, HH * DD, DD, DD, HH * DD, HH * DD, 0, 0, 0, 1, 0, 1);
        k_permute_h<<<(nH + ELT - 1) / ELT, ELT, 0, stream>>>(hlin, h);

        // hW = h @ We[k] : [16384,128]@[128,128]
        gemm(h, We + (size_t)k * DD * DD, hW,
             BB * HH * NB, DD, DD, DD, DD, DD, 0, 0, 0, 1, 0, 0);

        // e = hW @ h^T, batched over 16 (b,h)
        gemm(hW, h, e, NB, NB, DD, DD, DD, NB,
             (long long)NB * DD, (long long)NB * DD, (long long)NB * NB,
             BB * HH, 1, 0);

        // e = e + e^T
        k_esym<<<dim3(NB / 16, NB / 16, BB * HH), dim3(16, 16), 0, stream>>>(e);

        for (int br = 0; br < 2; ++br) {
            // masked softmax over axis=-2, post-scaled by adj
            k_masked_softmax<<<dim3(NB / 256, BB * HH), 256, 0, stream>>>(
                e, adjs[br], att);

            // z = h
            hipMemcpyAsync(z, h, SZ_H * sizeof(float),
                           hipMemcpyDeviceToDevice, stream);

            for (int hop = 0; hop <= k; ++hop) {
                // az = relu(att @ z), batched
                gemm(att, z, az, NB, DD, NB, NB, DD, DD,
                     (long long)NB * NB, (long long)NB * DD, (long long)NB * DD,
                     BB * HH, 0, 1);
                if (hop == 0)
                    k_beta<<<(BB * HH * NB + ELT - 1) / ELT, ELT, 0, stream>>>(
                        h, az, Wbw + (size_t)k * 2 * DD, Wbb + k, beta);
                k_zupdate<<<(nH + ELT - 1) / ELT, ELT, 0, stream>>>(h, az, beta, z);
            }

            // out_branch = relu(zcat) @ Wo[k]
            k_zcat_relu<<<(nH + ELT - 1) / ELT, ELT, 0, stream>>>(z, zcat);
            gemm(zcat, Wo + (size_t)k * HH * DD * DD, couts[br],
                 BB * NB, DD, HH * DD, HH * DD, DD, DD, 0, 0, 0, 1, 0, 0);
        }
        // c = c2 - c1
        k_combine<<<(nC + ELT - 1) / ELT, ELT, 0, stream>>>(c2buf, c1buf, c);
    }

    // masked mean pool + FC head
    k_pool<<<BB, DD, 0, stream>>>(c, valid, pooled);
    k_fc<<<(BB * DFCN + ELT - 1) / ELT, ELT, 0, stream>>>(
        pooled, fc0_w, fc0_b, f0, BB, DD, DFCN, 1);
    k_fc<<<(BB * DFCN + ELT - 1) / ELT, ELT, 0, stream>>>(
        f0, fc1_w, fc1_b, f1, BB, DFCN, DFCN, 1);
    k_fc<<<1, BB, 0, stream>>>(f1, fc2_w, fc2_b, (float*)d_out, BB, DFCN, 1, 2);
}
